// RelPosMultiHeadSelfAttention_88218628260744
// MI455X (gfx1250) — matrix operand order repaired and checked by reading, not compiled, not run on hardware
//
#include <hip/hip_runtime.h>
#include <hip/hip_bf16.h>

#define B_  8
#define T_  512
#define D_  1024
#define H_  16
#define DH_ 64
#define R_  1023  // 2T-1

typedef __attribute__((ext_vector_type(16))) __bf16 v16bf;
typedef __attribute__((ext_vector_type(8)))  float  v8f;

// ---------- helpers ----------
__device__ inline unsigned short f2bf_bits(float f) {
  unsigned u = __float_as_uint(f);
  u += 0x7FFFu + ((u >> 16) & 1u);   // round-to-nearest-even
  return (unsigned short)(u >> 16);
}

__device__ inline void st_out(float* p, float v)          { *p = v; }
__device__ inline void st_out(unsigned short* p, float v) { *p = f2bf_bits(v); }

__device__ inline int imin(int a, int b) { return a < b ? a : b; }

// ---------- WMMA GEMM (bf16 x bf16 -> f32 acc): C(MxN) = A(MxK) @ B^T + bias ----------
// A stored (M,K) row-major, B stored (N,K) row-major (both k-contiguous, bf16 bits).
// LDS tiles filled by CDNA5 async DMA (global_load_async_to_lds_b128), double
// buffered: issue tile i+1, s_wait_asynccnt 4 keeps it in flight while WMMA
// consumes tile i (async loads complete in order per wave).
#define BM 64
#define BN 64
#define BK 32
#define LSTR 40                       // padded LDS row stride (bf16 elems)
#define TILE_BYTES (BM * LSTR * 2)    // one 64x32 tile (padded)

template <typename OT>
__global__ __launch_bounds__(128) void gemm_wmma_async(
    const unsigned short* __restrict__ A, int lda, long long sAz,
    const unsigned short* __restrict__ Bm, int ldb, long long sBz,
    OT* __restrict__ C, int ldc, long long sCz,
    const float* __restrict__ bias, int M, int N, int K) {
  __shared__ __attribute__((aligned(16))) unsigned short As[2][BM * LSTR];
  __shared__ __attribute__((aligned(16))) unsigned short Bs[2][BN * LSTR];

  A  += (long long)blockIdx.z * sAz;
  Bm += (long long)blockIdx.z * sBz;
  C  += (long long)blockIdx.z * sCz;

  const int m0   = blockIdx.y * BM;
  const int n0   = blockIdx.x * BN;
  const int tid  = threadIdx.x;
  const int lane = tid & 31;        // wave32
  const int wv   = tid >> 5;        // 4 waves / block
  const int lm   = lane & 15;
  const int hi   = lane >> 4;

  v8f acc[4];
#pragma unroll
  for (int t = 0; t < 4; ++t)
    acc[t] = (v8f){0.f, 0.f, 0.f, 0.f, 0.f, 0.f, 0.f, 0.f};

  // staging map: thread -> (row, 16-col half); 128 thr * 32B = one 64x32 bf16 tile
  const int srow = tid >> 1;
  const int scol = (tid & 1) * 16;
  // clamp OOB tail rows to a valid row: garbage contributions land only in
  // output rows >= M / cols >= N, which the guarded store discards.
  const int garow = imin(m0 + srow, M - 1);
  const int gbrow = imin(n0 + srow, N - 1);
  const unsigned short* aSrcRow = A  + (size_t)garow * lda + scol;
  const unsigned short* bSrcRow = Bm + (size_t)gbrow * ldb + scol;
  // wave-relative LDS byte offsets (low 32 bits of the flat shared address)
  const unsigned aDst0 = (unsigned)(uintptr_t)&As[0][srow * LSTR + scol];
  const unsigned bDst0 = (unsigned)(uintptr_t)&Bs[0][srow * LSTR + scol];

  // issue one tile pair (4 async b128 ops per thread) into buffer `buf`
  auto issue = [&](int k0, int buf) {
    const unsigned short* ag = aSrcRow + k0;
    const unsigned short* bg = bSrcRow + k0;
    const unsigned ad = aDst0 + buf * TILE_BYTES;
    const unsigned bd = bDst0 + buf * TILE_BYTES;
    asm volatile(
        "global_load_async_to_lds_b128 %0, %2, off\n\t"
        "global_load_async_to_lds_b128 %0, %2, off offset:16\n\t"
        "global_load_async_to_lds_b128 %1, %3, off\n\t"
        "global_load_async_to_lds_b128 %1, %3, off offset:16"
        :
        : "v"(ad), "v"(bd), "v"(ag), "v"(bg)
        : "memory");
  };

  const int nsteps = K / BK;   // K is always a multiple of BK here
  issue(0, 0);                 // prologue: tile 0 -> buf 0

  for (int i = 0; i < nsteps; ++i) {
    const int buf = i & 1;
    const bool more = (i + 1) < nsteps;
    if (more) {
      issue((i + 1) * BK, buf ^ 1);             // keep next tile in flight
      __builtin_prefetch(aSrcRow + (size_t)(i + 2) * BK, 0, 2);
      asm volatile("s_wait_asynccnt 4" ::: "memory");  // tile i resident
    } else {
      asm volatile("s_wait_asynccnt 0" ::: "memory");
    }
    __syncthreads();   // all waves' tile i visible in LDS

    // ---- A fragment: 16x32 bf16, rows m0 + 16*wv .. +15 ----
    union { v16bf v; uint4 q[2]; } af;
    const unsigned short* ap = &As[buf][(wv * 16 + lm) * LSTR];
    af.q[0] = *reinterpret_cast<const uint4*>(ap + 8 * hi);
    af.q[1] = *reinterpret_cast<const uint4*>(ap + 16 + 8 * hi);

#pragma unroll
    for (int t = 0; t < 4; ++t) {
      union { v16bf v; uint4 q[2]; } bfr;
      const unsigned short* bp = &Bs[buf][(t * 16 + lm) * LSTR];
      bfr.q[0] = *reinterpret_cast<const uint4*>(bp + 8 * hi);
      bfr.q[1] = *reinterpret_cast<const uint4*>(bp + 16 + 8 * hi);
      acc[t] = __builtin_amdgcn_wmma_f32_16x16x32_bf16(
          false, af.v, false, bfr.v, (short)0, acc[t], false, false);
    }
    __syncthreads();   // tile i free for reuse before iteration i+1 re-issues it
  }

  // ---- store C (+bias) ----
#pragma unroll
  for (int t = 0; t < 4; ++t) {
    const int c = n0 + t * 16 + lm;
    float badd = (bias && c < N) ? bias[c] : 0.f;
#pragma unroll
    for (int v2 = 0; v2 < 8; ++v2) {
      const int r = m0 + wv * 16 + hi * 8 + v2;
      if (r < M && c < N)
        st_out(&C[(size_t)r * ldc + c], acc[t][v2] + badd);
    }
  }
}

// ---------- fp32 -> bf16 bulk convert (vectorized) ----------
__global__ void cvt_bf16_kernel(const float* __restrict__ s,
                                unsigned short* __restrict__ d, int n4) {
  int i = blockIdx.x * blockDim.x + threadIdx.x;
  if (i < n4) {
    float4 f = reinterpret_cast<const float4*>(s)[i];
    uint2 o;
    o.x = (unsigned)f2bf_bits(f.x) | ((unsigned)f2bf_bits(f.y) << 16);
    o.y = (unsigned)f2bf_bits(f.z) | ((unsigned)f2bf_bits(f.w) << 16);
    reinterpret_cast<uint2*>(d)[i] = o;
  }
}

// ---------- relative sinusoidal PE table, written directly as bf16 ----------
__global__ void pe_kernel(unsigned short* __restrict__ PE) {
  const int r = blockIdx.x;               // [0, R_)
  const float p = 511.f - (float)r;       // pos[r] (T == MAX_LEN -> full table)
  for (int i = threadIdx.x; i < D_ / 2; i += blockDim.x) {
    const float invf = __expf(-(float)i * (9.210340371976184f / 512.f)); // 10000^(-2i/D)
    const float a = p * invf;
    PE[(size_t)r * D_ + 2 * i]     = f2bf_bits(sinf(a));
    PE[(size_t)r * D_ + 2 * i + 1] = f2bf_bits(cosf(a));
  }
}

__global__ void biasuv_kernel(const float* __restrict__ bq, const float* __restrict__ u,
                              const float* __restrict__ v, float* __restrict__ bu,
                              float* __restrict__ bv) {
  int i = blockIdx.x * blockDim.x + threadIdx.x;
  if (i < D_) { bu[i] = bq[i] + u[i]; bv[i] = bq[i] + v[i]; }
}

// ---------- V head transpose: Vt[b,h,d,t] = V[b,t,h*64+d] ----------
__global__ __launch_bounds__(256) void vtrans_kernel(const unsigned short* __restrict__ Vb,
                                                     unsigned short* __restrict__ Vt) {
  const int bh = blockIdx.y;              // b*H + h
  const int b = bh >> 4, h = bh & 15;
  const int t = blockIdx.x * 64 + (threadIdx.x & 63);
  const int d0 = (threadIdx.x >> 6) * 16; // 4 groups of 16 d's
#pragma unroll
  for (int dd = 0; dd < 16; ++dd) {
    const int d = d0 + dd;
    Vt[((size_t)bh * DH_ + d) * T_ + t] = Vb[((size_t)b * T_ + t) * D_ + h * 64 + d];
  }
}

// ---------- combine (skewed E term), scale, softmax -> bf16 probs ----------
__global__ __launch_bounds__(256) void softmax_combine(
    const float* __restrict__ sK, const float* __restrict__ sE,
    unsigned short* __restrict__ P) {
  const int q = blockIdx.x;
  const int h = blockIdx.y;
  const float* rk = sK + ((size_t)h * T_ + q) * T_;
  const float* re = sE + ((size_t)h * T_ + q) * R_ + (T_ - 1 - q); // re[k]=sE[q][k-q+T-1]
  unsigned short* po = P + ((size_t)h * T_ + q) * T_;

  const int t0 = threadIdx.x;
  float l0 = (rk[t0] + re[t0]) * 0.125f;                 // 1/sqrt(64)
  float l1 = (rk[t0 + 256] + re[t0 + 256]) * 0.125f;

  __shared__ float red[8];
  __shared__ float red2[8];
  const int lane = t0 & 31, wv = t0 >> 5;

  float m = fmaxf(l0, l1);
#pragma unroll
  for (int off = 16; off > 0; off >>= 1) m = fmaxf(m, __shfl_xor(m, off, 32));
  if (lane == 0) red[wv] = m;
  __syncthreads();
  float mm = red[0];
#pragma unroll
  for (int i = 1; i < 8; ++i) mm = fmaxf(mm, red[i]);

  float e0 = __expf(l0 - mm), e1 = __expf(l1 - mm);
  float s = e0 + e1;
#pragma unroll
  for (int off = 16; off > 0; off >>= 1) s += __shfl_xor(s, off, 32);
  if (lane == 0) red2[wv] = s;
  __syncthreads();
  float ss = 0.f;
#pragma unroll
  for (int i = 0; i < 8; ++i) ss += red2[i];
  const float inv = 1.f / ss;

  po[t0]       = f2bf_bits(e0 * inv);
  po[t0 + 256] = f2bf_bits(e1 * inv);
}

// ---------- host driver ----------
extern "C" void kernel_launch(void* const* d_in, const int* in_sizes, int n_in,
                              void* d_out, int out_size, void* d_ws, size_t ws_size,
                              hipStream_t stream) {
  (void)in_sizes; (void)n_in; (void)out_size; (void)ws_size;
  const float* x  = (const float*)d_in[0];
  const float* Wq = (const float*)d_in[1];
  const float* bq = (const float*)d_in[2];
  const float* Wk = (const float*)d_in[3];
  const float* bk = (const float*)d_in[4];
  const float* Wv = (const float*)d_in[5];
  const float* bv = (const float*)d_in[6];
  const float* Wp = (const float*)d_in[7];
  const float* bp = (const float*)d_in[8];
  const float* Wo = (const float*)d_in[9];
  const float* bo = (const float*)d_in[10];
  const float* u  = (const float*)d_in[11];
  const float* v  = (const float*)d_in[12];
  float* out = (float*)d_out;

  char* wsp = (char*)d_ws;
  auto alloc = [&](size_t bytes) {
    char* p = wsp;
    wsp += (bytes + 255) & ~(size_t)255;
    return p;
  };
  const size_t BTD = (size_t)B_ * T_ * D_;
  unsigned short* xb  = (unsigned short*)alloc(BTD * 2);
  unsigned short* Wqb = (unsigned short*)alloc((size_t)D_ * D_ * 2);
  unsigned short* Wkb = (unsigned short*)alloc((size_t)D_ * D_ * 2);
  unsigned short* Wvb = (unsigned short*)alloc((size_t)D_ * D_ * 2);
  unsigned short* Wpb = (unsigned short*)alloc((size_t)D_ * D_ * 2);
  unsigned short* Wob = (unsigned short*)alloc((size_t)D_ * D_ * 2);
  unsigned short* PEb = (unsigned short*)alloc((size_t)R_ * D_ * 2);
  float*          bu  = (float*)alloc((size_t)D_ * 4);
  float*          bvv = (float*)alloc((size_t)D_ * 4);
  unsigned short* Qu  = (unsigned short*)alloc(BTD * 2);
  unsigned short* Qv  = (unsigned short*)alloc(BTD * 2);
  unsigned short* Kb  = (unsigned short*)alloc(BTD * 2);
  unsigned short* Vb  = (unsigned short*)alloc(BTD * 2);
  unsigned short* Vt  = (unsigned short*)alloc(BTD * 2);
  unsigned short* E   = (unsigned short*)alloc((size_t)R_ * D_ * 2);
  unsigned short* Oh  = (unsigned short*)alloc(BTD * 2);
  float*          sK  = (float*)alloc((size_t)H_ * T_ * T_ * 4);
  float*          sE  = (float*)alloc((size_t)H_ * T_ * R_ * 4);
  unsigned short* P   = (unsigned short*)alloc((size_t)H_ * T_ * T_ * 2);

  const dim3 blk(128);

  // one-time bf16 conversions + PE table + fused biases
  cvt_bf16_kernel<<<dim3((unsigned)(BTD / 4 / 256)), dim3(256), 0, stream>>>(x, xb, (int)(BTD / 4));
  cvt_bf16_kernel<<<dim3(D_ * D_ / 4 / 256), dim3(256), 0, stream>>>(Wq, Wqb, D_ * D_ / 4);
  cvt_bf16_kernel<<<dim3(D_ * D_ / 4 / 256), dim3(256), 0, stream>>>(Wk, Wkb, D_ * D_ / 4);
  cvt_bf16_kernel<<<dim3(D_ * D_ / 4 / 256), dim3(256), 0, stream>>>(Wv, Wvb, D_ * D_ / 4);
  cvt_bf16_kernel<<<dim3(D_ * D_ / 4 / 256), dim3(256), 0, stream>>>(Wp, Wpb, D_ * D_ / 4);
  cvt_bf16_kernel<<<dim3(D_ * D_ / 4 / 256), dim3(256), 0, stream>>>(Wo, Wob, D_ * D_ / 4);
  pe_kernel<<<dim3(R_), dim3(256), 0, stream>>>(PEb);
  biasuv_kernel<<<dim3(4), dim3(256), 0, stream>>>(bq, u, v, bu, bvv);

  // Projections: (B*T,D) @ W^T + bias -> bf16
  const dim3 gproj(D_ / 64, (B_ * T_) / 64);
  gemm_wmma_async<unsigned short><<<gproj, blk, 0, stream>>>(
      xb, D_, 0, Wqb, D_, 0, Qu, D_, 0, bu, B_ * T_, D_, D_);
  gemm_wmma_async<unsigned short><<<gproj, blk, 0, stream>>>(
      xb, D_, 0, Wqb, D_, 0, Qv, D_, 0, bvv, B_ * T_, D_, D_);
  gemm_wmma_async<unsigned short><<<gproj, blk, 0, stream>>>(
      xb, D_, 0, Wkb, D_, 0, Kb, D_, 0, bk, B_ * T_, D_, D_);
  gemm_wmma_async<unsigned short><<<gproj, blk, 0, stream>>>(
      xb, D_, 0, Wvb, D_, 0, Vb, D_, 0, bv, B_ * T_, D_, D_);
  // E = PE @ Wp^T + bp  (M = R_ = 1023, tail-clamped)
  gemm_wmma_async<unsigned short><<<dim3(D_ / 64, (R_ + 63) / 64), blk, 0, stream>>>(
      PEb, D_, 0, Wpb, D_, 0, E, D_, 0, bp, R_, D_, D_);
  // Vt[b,h,d,t] for the P@V GEMM (turns it into the same NT form)
  vtrans_kernel<<<dim3(T_ / 64, B_ * H_), dim3(256), 0, stream>>>(Vb, Vt);

  for (int b = 0; b < B_; ++b) {
    const unsigned short* Qub = Qu + (size_t)b * T_ * D_;
    const unsigned short* Qvb = Qv + (size_t)b * T_ * D_;
    const unsigned short* Kbb = Kb + (size_t)b * T_ * D_;
    const unsigned short* Vtb = Vt + (size_t)b * H_ * DH_ * T_;

    // sK[h] = Qu_h (T x dh) @ K_h^T   (heads batched in grid.z)
    gemm_wmma_async<float><<<dim3(T_ / 64, T_ / 64, H_), blk, 0, stream>>>(
        Qub, D_, 64, Kbb, D_, 64, sK, T_, (long long)T_ * T_, nullptr, T_, T_, DH_);
    // sE[h] = Qv_h (T x dh) @ E_h^T   (N = R_ = 1023, tail-clamped)
    gemm_wmma_async<float><<<dim3((R_ + 63) / 64, T_ / 64, H_), blk, 0, stream>>>(
        Qvb, D_, 64, E, D_, 64, sE, R_, (long long)T_ * R_, nullptr, T_, R_, DH_);
    // skew + scale + softmax -> bf16 probs
    softmax_combine<<<dim3(T_, H_), dim3(256), 0, stream>>>(sK, sE, P);
    // O_h = P_h (T x T) @ Vt_h^T (Vt is (dh,T): NT form), head-interleaved into Oh
    gemm_wmma_async<unsigned short><<<dim3(1, T_ / 64, H_), blk, 0, stream>>>(
        P, T_, (long long)T_ * T_, Vtb, T_, (long long)DH_ * T_,
        Oh + (size_t)b * T_ * D_, D_, 64, nullptr, T_, DH_, T_);
  }

  // out = Oh @ Wo^T + bo  (fp32 output)
  gemm_wmma_async<float><<<gproj, blk, 0, stream>>>(
      Oh, D_, 0, Wob, D_, 0, out, D_, 0, bo, B_ * T_, D_, D_);
}
